// CausalSelfAttention_3994319585445
// MI455X (gfx1250) — matrix operand order repaired and checked by reading, not compiled, not run on hardware
//
#include <hip/hip_runtime.h>

// ---------------------------------------------------------------------------
// CDNA5 (gfx1250) causal self-attention, bf16 WMMA pipeline.
//   - register-blocked GEMMs: 8 v_wmma per wave per K-step (32x64 per wave)
//   - double-buffered GLOBAL_LOAD_ASYNC_TO_LDS staging (ASYNCcnt) when available
//   - flash attention with online softmax, all matmuls on v_wmma_f32_16x16x32_bf16
// ---------------------------------------------------------------------------

typedef __attribute__((ext_vector_type(16))) __bf16 v16bf;
typedef __attribute__((ext_vector_type(8)))  __bf16 bf16x8;
typedef __attribute__((ext_vector_type(8)))  float  v8f;
typedef __attribute__((__vector_size__(4 * sizeof(int)))) int v4i;

#define AS1 __attribute__((address_space(1)))
#define AS3 __attribute__((address_space(3)))

#define TT   2048
#define DD   768
#define NH   12
#define HD   64

#if __has_builtin(__builtin_amdgcn_global_load_async_to_lds_b128)
#define HAVE_ASYNC 1
#else
#define HAVE_ASYNC 0
#endif

__device__ __forceinline__ void wait_async0() {
#if HAVE_ASYNC
#if __has_builtin(__builtin_amdgcn_s_wait_asynccnt)
    __builtin_amdgcn_s_wait_asynccnt(0);
#else
    asm volatile("s_wait_asynccnt 0" ::: "memory");
#endif
#endif
}

#if HAVE_ASYNC
__device__ __forceinline__ void async_b128(const __bf16* g, __bf16* l) {
    __builtin_amdgcn_global_load_async_to_lds_b128(
        (AS1 v4i*)(void*)g,
        (AS3 v4i*)(void*)l,
        0, 0);
}
#endif

// copy 8 bf16 global->LDS (async if available, else sync vector copy)
__device__ __forceinline__ void stage8(const __bf16* g, __bf16* l) {
#if HAVE_ASYNC
    async_b128(g, l);
#else
    *(bf16x8*)l = *(const bf16x8*)g;
#endif
}

// float -> bf16, round-to-nearest-even, integer-only (always selectable)
__device__ __forceinline__ __bf16 f2bf(float f) {
    union { float f; unsigned u; } v; v.f = f;
    unsigned r = v.u + 0x7FFFu + ((v.u >> 16) & 1u);
    unsigned short h = (unsigned short)(r >> 16);
    return __builtin_bit_cast(__bf16, h);
}

// A-fragment (16x32 bf16, M rows striped over lanes; ISA 7.12.2 16-bit A map)
__device__ __forceinline__ v16bf ldsA(const __bf16* base, int ld, int koff, int lane) {
    int m = lane & 15, h = (lane >> 4) & 1;
    v16bf f;
#pragma unroll
    for (int v = 0; v < 8; ++v) {
        int k = (v < 4) ? (h * 8 + 2 * v) : (16 + h * 8 + 2 * (v - 4));
        const __bf16* p = base + (size_t)m * ld + koff + k;
        f[2 * v]     = p[0];
        f[2 * v + 1] = p[1];
    }
    return f;
}

// B-fragment (32x16 bf16) read from Bt[n][k] storage (ISA 7.12.2 16-bit B map)
__device__ __forceinline__ v16bf ldsB(const __bf16* base, int ld, int koff, int lane) {
    int n = lane & 15, h = (lane >> 4) & 1;
    v16bf f;
#pragma unroll
    for (int v = 0; v < 8; ++v) {
        int k = koff + ((v < 4) ? (h * 8 + 2 * v) : (16 + h * 8 + 2 * (v - 4)));
        const __bf16* p = base + (size_t)n * ld + k;
        f[2 * v]     = p[0];
        f[2 * v + 1] = p[1];
    }
    return f;
}

__device__ __forceinline__ v8f wmma_bf16(v16bf a, v16bf b, v8f c) {
    return __builtin_amdgcn_wmma_f32_16x16x32_bf16(false, a, false, b,
                                                   (short)0, c, false, false);
}

// ---------------------------------------------------------------------------
// 1) f32 -> bf16 cast
// ---------------------------------------------------------------------------
__global__ void cast_bf16_kernel(const float* __restrict__ in,
                                 __bf16* __restrict__ out, int n) {
    int i = blockIdx.x * blockDim.x + threadIdx.x;
    if (i < n) out[i] = f2bf(in[i]);
}

// ---------------------------------------------------------------------------
// Register-blocked GEMM core:  C[m,n] = sum_k A[m,k] * W[n,k]  (A @ W^T)
// Block tile 64x128, 4 waves (2x2), each wave: 32x64 = 2x4 WMMA tiles.
// Double-buffered async tile staging.
// ---------------------------------------------------------------------------
#define GBM 64
#define GBN 128
#define GBK 32
#define AS_ELEMS (GBM * GBK)   // 2048
#define WS_ELEMS (GBN * GBK)   // 4096

__device__ __forceinline__ void tile_issue(const __bf16* __restrict__ A,
                                           const __bf16* __restrict__ W,
                                           int m0, int n0, int k0,
                                           __bf16* As, __bf16* Ws, int tid) {
    // As: 64 rows x 32 cols -> thread: row = tid/2, 16 cols
    {
        int r = tid >> 1, c = (tid & 1) * 16;
        const __bf16* g = A + (size_t)(m0 + r) * DD + k0 + c;
        __bf16* l = &As[r * GBK + c];
        stage8(g, l);
        stage8(g + 8, l + 8);
    }
    // Ws: 128 rows x 32 cols -> thread: one row, 32 cols
    {
        const __bf16* g = W + (size_t)(n0 + tid) * DD + k0;
        __bf16* l = &Ws[tid * GBK];
        stage8(g,      l);
        stage8(g + 8,  l + 8);
        stage8(g + 16, l + 16);
        stage8(g + 24, l + 24);
    }
}

__device__ __forceinline__ void gemm_core(const __bf16* __restrict__ A,
                                          const __bf16* __restrict__ W,
                                          int m0, int n0, int tid,
                                          __bf16* AsBuf, __bf16* WsBuf,
                                          v8f acc[2][4]) {
    const int lane = tid & 31;
    const int wave = tid >> 5;
    const int wy = wave >> 1, wx = wave & 1;

#pragma unroll
    for (int mi = 0; mi < 2; ++mi)
#pragma unroll
        for (int ni = 0; ni < 4; ++ni)
#pragma unroll
            for (int i = 0; i < 8; ++i) acc[mi][ni][i] = 0.0f;

    tile_issue(A, W, m0, n0, 0, AsBuf, WsBuf, tid);

    for (int k0 = 0; k0 < DD; k0 += GBK) {
        const int b = (k0 >> 5) & 1;
        wait_async0();
        __syncthreads();
        if (k0 + GBK < DD)
            tile_issue(A, W, m0, n0, k0 + GBK,
                       AsBuf + (b ^ 1) * AS_ELEMS, WsBuf + (b ^ 1) * WS_ELEMS, tid);

        const __bf16* As = AsBuf + b * AS_ELEMS;
        const __bf16* Ws = WsBuf + b * WS_ELEMS;

        v16bf fa[2], fb[4];
#pragma unroll
        for (int mi = 0; mi < 2; ++mi)
            fa[mi] = ldsA(As + (wy * 32 + mi * 16) * GBK, GBK, 0, lane);
#pragma unroll
        for (int ni = 0; ni < 4; ++ni)
            fb[ni] = ldsB(Ws + (wx * 64 + ni * 16) * GBK, GBK, 0, lane);

#pragma unroll
        for (int mi = 0; mi < 2; ++mi)
#pragma unroll
            for (int ni = 0; ni < 4; ++ni)
                acc[mi][ni] = wmma_bf16(fa[mi], fb[ni], acc[mi][ni]);
        __syncthreads();
    }
}

// ---------------------------------------------------------------------------
// 2) Fused QKV GEMM. grid = (DD/128, TT/64, 3); block = 128
// ---------------------------------------------------------------------------
__global__ void qkv_gemm_kernel(const __bf16* __restrict__ X,
                                const __bf16* __restrict__ Wq,
                                const __bf16* __restrict__ Wk,
                                const __bf16* __restrict__ Wv,
                                const float*  __restrict__ vi,
                                const float*  __restrict__ lamb,
                                float* __restrict__ qf,
                                float* __restrict__ kf,
                                __bf16* __restrict__ vb) {
    __shared__ __bf16 AsBuf[2 * AS_ELEMS];
    __shared__ __bf16 WsBuf[2 * WS_ELEMS];

    const int which = blockIdx.z;
    const __bf16* W = (which == 0) ? Wq : (which == 1) ? Wk : Wv;
    const int m0 = blockIdx.y * GBM;
    const int n0 = blockIdx.x * GBN;
    const int tid = threadIdx.x;

    v8f acc[2][4];
    gemm_core(X, W, m0, n0, tid, AsBuf, WsBuf, acc);

    const int lane = tid & 31;
    const int wave = tid >> 5;
    const int wy = wave >> 1, wx = wave & 1;
    const int col = lane & 15, hh = lane >> 4;

#pragma unroll
    for (int mi = 0; mi < 2; ++mi)
#pragma unroll
        for (int r = 0; r < 8; ++r) {
            int row = m0 + wy * 32 + mi * 16 + r + 8 * hh;
#pragma unroll
            for (int ni = 0; ni < 4; ++ni) {
                int coln = n0 + wx * 64 + ni * 16 + col;
                float a = acc[mi][ni][r];
                if (which == 0)      qf[(size_t)row * DD + coln] = a;
                else if (which == 1) kf[(size_t)row * DD + coln] = a;
                else {
                    float lam = lamb[0];
                    float v = (1.0f - lam) * a + lam * vi[(size_t)row * DD + coln];
                    vb[(size_t)row * DD + coln] = f2bf(v);
                }
            }
        }
}

// ---------------------------------------------------------------------------
// 3) RMS-norm + RoPE per (t, head); one wave each.
//    grid = (T*NH/4, 2); block = 128.  grid.y: 0 -> q, 1 -> k
// ---------------------------------------------------------------------------
__global__ void norm_rope_kernel(const float* __restrict__ qf,
                                 const float* __restrict__ kf,
                                 __bf16* __restrict__ qh,
                                 __bf16* __restrict__ kh) {
    const int lane = threadIdx.x & 31;
    const int wid  = blockIdx.x * 4 + (threadIdx.x >> 5);   // 0 .. T*NH-1
    const int t  = wid / NH;
    const int hd = wid % NH;

    const float* src = blockIdx.y ? kf : qf;
    __bf16*      dst = blockIdx.y ? kh : qh;

    const float* row = src + (size_t)t * DD + hd * HD;
    float x1 = row[lane];
    float x2 = row[lane + 32];

    float ss = x1 * x1 + x2 * x2;
#pragma unroll
    for (int off = 1; off < 32; off <<= 1) ss += __shfl_xor(ss, off, 32);
    float scale = rsqrtf(ss * (1.0f / 64.0f) + 1.1920929e-7f);
    x1 *= scale; x2 *= scale;

    // inv_freq[i] = 10000^(-i/32), freq = t * inv_freq
    float fr = (float)t * __powf(10000.0f, -((float)lane) * (1.0f / 32.0f));
    float c = __cosf(fr), s = __sinf(fr);
    float y1 =  x1 * c + x2 * s;
    float y2 = -x1 * s + x2 * c;

    __bf16* drow = dst + (size_t)t * DD + hd * HD;
    drow[lane]      = f2bf(y1);
    drow[lane + 32] = f2bf(y2);
}

// ---------------------------------------------------------------------------
// 4) Flash attention (causal).  grid = (T/64, NH); block = 128 (4 waves).
//    Wave w owns q rows [q0+16w, q0+16w+16). 32-key tiles, online softmax.
// ---------------------------------------------------------------------------
__global__ void flash_attn_kernel(const __bf16* __restrict__ qh,
                                  const __bf16* __restrict__ kh,
                                  const __bf16* __restrict__ vb,
                                  __bf16* __restrict__ ob) {
    __shared__ __bf16 Qs[64 * 64];     // q block, row-major [qrow][d]
    __shared__ __bf16 Ks[32 * 64];     // key tile, row-major [key][d]
    __shared__ __bf16 Vt[64 * 32];     // V^T tile [d][key]  (== Bt for P@V)
    __shared__ __bf16 Ps[4][16 * 32];  // per-wave probabilities [m][key]

    const int h    = blockIdx.y;
    const int q0   = blockIdx.x * 64;
    const int lane = threadIdx.x & 31;
    const int w    = threadIdx.x >> 5;
    const int wq   = q0 + w * 16;
    const int tid  = threadIdx.x;

    // load Q block (64 x 64), async staged
    {
        int r = tid >> 1, cb = (tid & 1) * 32;
        const __bf16* src = qh + (size_t)(q0 + r) * DD + h * HD + cb;
        __bf16* dst = &Qs[r * 64 + cb];
#pragma unroll
        for (int v = 0; v < 4; ++v)
            stage8(src + v * 8, dst + v * 8);
    }
    wait_async0();
    __syncthreads();

    const v16bf qf0 = ldsA(Qs + w * 16 * 64, 64, 0,  lane);
    const v16bf qf1 = ldsA(Qs + w * 16 * 64, 64, 32, lane);

    float mstat[8], lstat[8];
    v8f acc[4];
#pragma unroll
    for (int r = 0; r < 8; ++r) { mstat[r] = -1e30f; lstat[r] = 0.0f; }
#pragma unroll
    for (int c = 0; c < 4; ++c)
#pragma unroll
        for (int r = 0; r < 8; ++r) acc[c][r] = 0.0f;

    const int half = lane >> 4, nlo = lane & 15;

    for (int kc = 0; kc < q0 + 64; kc += 32) {
        __syncthreads();
        // K tile (32x64): async staged. V tile: loaded + transposed to Vt.
        {
            int r = tid >> 2, cb = (tid & 3) * 16;
            const __bf16* ksrc = kh + (size_t)(kc + r) * DD + h * HD + cb;
            stage8(ksrc,     &Ks[r * 64 + cb]);
            stage8(ksrc + 8, &Ks[r * 64 + cb + 8]);
            const __bf16* vsrc = vb + (size_t)(kc + r) * DD + h * HD + cb;
            bf16x8 v0 = *(const bf16x8*)&vsrc[0];
            bf16x8 v1 = *(const bf16x8*)&vsrc[8];
#pragma unroll
            for (int i = 0; i < 8; ++i) {
                Vt[(cb + i) * 32 + r]     = v0[i];
                Vt[(cb + 8 + i) * 32 + r] = v1[i];
            }
        }
        wait_async0();
        __syncthreads();

        // S = Q @ K^T  (two 16x16 tiles, each K-dim 64 = 2 WMMA)
        v8f s[2];
#pragma unroll
        for (int j = 0; j < 2; ++j) {
            v16bf kfa = ldsB(Ks + j * 16 * 64, 64, 0,  lane);
            v16bf kfb = ldsB(Ks + j * 16 * 64, 64, 32, lane);
            v8f tacc;
#pragma unroll
            for (int i = 0; i < 8; ++i) tacc[i] = 0.0f;
            tacc = wmma_bf16(qf0, kfa, tacc);
            tacc = wmma_bf16(qf1, kfb, tacc);
            s[j] = tacc;
        }

        // online softmax (rows striped r + 8*half; reduce across 16-lane half)
#pragma unroll
        for (int r = 0; r < 8; ++r) {
            int row = wq + r + 8 * half;
            int c0 = kc + nlo, c1 = kc + 16 + nlo;
            bool msk0 = c0 > row, msk1 = c1 > row;
            float s0 = msk0 ? -1e30f : s[0][r] * 0.125f;
            float s1 = msk1 ? -1e30f : s[1][r] * 0.125f;

            float rmax = fmaxf(s0, s1);
#pragma unroll
            for (int off = 1; off < 16; off <<= 1)
                rmax = fmaxf(rmax, __shfl_xor(rmax, off, 16));

            float nm    = fmaxf(mstat[r], rmax);
            float alpha = __expf(mstat[r] - nm);
            float p0 = msk0 ? 0.0f : __expf(s0 - nm);
            float p1 = msk1 ? 0.0f : __expf(s1 - nm);

            float rs = p0 + p1;
#pragma unroll
            for (int off = 1; off < 16; off <<= 1)
                rs += __shfl_xor(rs, off, 16);

            lstat[r] = lstat[r] * alpha + rs;
            mstat[r] = nm;
#pragma unroll
            for (int c = 0; c < 4; ++c) acc[c][r] *= alpha;

            Ps[w][(r + 8 * half) * 32 + nlo]      = f2bf(p0);
            Ps[w][(r + 8 * half) * 32 + 16 + nlo] = f2bf(p1);
        }
        __syncthreads();

        // O += P @ V  (4 output 16x16 tiles across head dim)
        v16bf pf = ldsA(Ps[w], 32, 0, lane);
#pragma unroll
        for (int c = 0; c < 4; ++c) {
            v16bf vf = ldsB(Vt + c * 16 * 32, 32, 0, lane);
            acc[c] = wmma_bf16(pf, vf, acc[c]);
        }
    }

    // normalize and store bf16
#pragma unroll
    for (int r = 0; r < 8; ++r) {
        float inv = 1.0f / lstat[r];
        __bf16* dst = ob + (size_t)(wq + r + 8 * half) * DD + h * HD;
#pragma unroll
        for (int c = 0; c < 4; ++c)
            dst[c * 16 + nlo] = f2bf(acc[c][r] * inv);
    }
}

// ---------------------------------------------------------------------------
// 5) Final projection: out[m,n] = sum_k O[m,k] * Wp[n,k]  (f32 out)
//    grid = (DD/128, TT/64); block = 128
// ---------------------------------------------------------------------------
__global__ void proj_gemm_kernel(const __bf16* __restrict__ A,
                                 const __bf16* __restrict__ W,
                                 float* __restrict__ C) {
    __shared__ __bf16 AsBuf[2 * AS_ELEMS];
    __shared__ __bf16 WsBuf[2 * WS_ELEMS];

    const int m0 = blockIdx.y * GBM;
    const int n0 = blockIdx.x * GBN;
    const int tid = threadIdx.x;

    v8f acc[2][4];
    gemm_core(A, W, m0, n0, tid, AsBuf, WsBuf, acc);

    const int lane = tid & 31;
    const int wave = tid >> 5;
    const int wy = wave >> 1, wx = wave & 1;
    const int col = lane & 15, hh = lane >> 4;

#pragma unroll
    for (int mi = 0; mi < 2; ++mi)
#pragma unroll
        for (int r = 0; r < 8; ++r) {
            int row = m0 + wy * 32 + mi * 16 + r + 8 * hh;
#pragma unroll
            for (int ni = 0; ni < 4; ++ni)
                C[(size_t)row * DD + n0 + wx * 64 + ni * 16 + col] = acc[mi][ni][r];
        }
}

// ---------------------------------------------------------------------------
// launch
// ---------------------------------------------------------------------------
extern "C" void kernel_launch(void* const* d_in, const int* in_sizes, int n_in,
                              void* d_out, int out_size, void* d_ws, size_t ws_size,
                              hipStream_t stream) {
    (void)in_sizes; (void)n_in; (void)out_size; (void)ws_size;

    const float* x    = (const float*)d_in[0];
    const float* vi   = (const float*)d_in[1];
    const float* Wq   = (const float*)d_in[2];
    const float* Wk   = (const float*)d_in[3];
    const float* Wv   = (const float*)d_in[4];
    const float* Wp   = (const float*)d_in[5];
    const float* lamb = (const float*)d_in[6];
    float* out = (float*)d_out;

    const size_t NX = (size_t)TT * DD;   // 1,572,864
    const size_t NW = (size_t)DD * DD;   //   589,824

    unsigned char* ws = (unsigned char*)d_ws;
    size_t off = 0;
    auto alloc = [&](size_t bytes) {
        void* p = ws + off;
        off = (off + bytes + 255) & ~(size_t)255;
        return p;
    };
    __bf16* xb  = (__bf16*)alloc(NX * 2);
    __bf16* wqb = (__bf16*)alloc(NW * 2);
    __bf16* wkb = (__bf16*)alloc(NW * 2);
    __bf16* wvb = (__bf16*)alloc(NW * 2);
    __bf16* wpb = (__bf16*)alloc(NW * 2);
    float*  qf  = (float*) alloc(NX * 4);
    float*  kf  = (float*) alloc(NX * 4);
    __bf16* vbb = (__bf16*)alloc(NX * 2);
    __bf16* qhb = (__bf16*)alloc(NX * 2);
    __bf16* khb = (__bf16*)alloc(NX * 2);
    __bf16* obb = (__bf16*)alloc(NX * 2);

    // 1) casts
    cast_bf16_kernel<<<(int)((NX + 255) / 256), 256, 0, stream>>>(x,  xb,  (int)NX);
    cast_bf16_kernel<<<(int)((NW + 255) / 256), 256, 0, stream>>>(Wq, wqb, (int)NW);
    cast_bf16_kernel<<<(int)((NW + 255) / 256), 256, 0, stream>>>(Wk, wkb, (int)NW);
    cast_bf16_kernel<<<(int)((NW + 255) / 256), 256, 0, stream>>>(Wv, wvb, (int)NW);
    cast_bf16_kernel<<<(int)((NW + 255) / 256), 256, 0, stream>>>(Wp, wpb, (int)NW);

    // 2) fused QKV projections (+ v blend)
    qkv_gemm_kernel<<<dim3(DD / GBN, TT / GBM, 3), 128, 0, stream>>>(
        xb, wqb, wkb, wvb, vi, lamb, qf, kf, vbb);

    // 3) RMS-norm + RoPE for q and k
    norm_rope_kernel<<<dim3((TT * NH) / 4, 2), 128, 0, stream>>>(qf, kf, qhb, khb);

    // 4) causal flash attention
    flash_attn_kernel<<<dim3(TT / 64, NH), 128, 0, stream>>>(qhb, khb, vbb, obb);

    // 5) output projection
    proj_gemm_kernel<<<dim3(DD / GBN, TT / GBM), 128, 0, stream>>>(obb, wpb, out);
}
